// DFFN_14010183320269
// MI455X (gfx1250) — matrix operand order repaired and checked by reading, not compiled, not run on hardware
//
#include <hip/hip_runtime.h>
#include <hip/hip_bf16.h>

typedef __attribute__((ext_vector_type(16))) __bf16 v16bf;
typedef __attribute__((ext_vector_type(8)))  float  v8f;

#define BATCH 2
#define DIM   384
#define HIDN  384
#define HSZ   256
#define WSZ   256
#define HW    65536
#define NBLK  1024   // (256/8)*(256/8)

union Frag16 { uint4 u[2]; v16bf v; };

__device__ __forceinline__ float bf2f(unsigned short s) {
    return __uint_as_float(((unsigned int)s) << 16);
}
// Packed f32x2 -> bf16x2 via hardware v_cvt_pk_bf16_f32 (RNE).
__device__ __forceinline__ unsigned int pack2(float lo, float hi) {
    unsigned int u;
    asm("v_cvt_pk_bf16_f32 %0, %1, %2" : "=v"(u) : "v"(lo), "v"(hi));
    return u;
}
__device__ __forceinline__ unsigned short f2bf(float f) {
    return (unsigned short)(pack2(f, 0.f) & 0xFFFFu);
}
__device__ __forceinline__ v8f wmma_bf16(const Frag16& a, const Frag16& b, v8f c) {
    return __builtin_amdgcn_wmma_f32_16x16x32_bf16(false, a.v, false, b.v,
                                                   (short)0, c, false, false);
}
// Async global->LDS copy of 16 bytes (ASYNCcnt-tracked).
__device__ __forceinline__ void async_g2l_b128(unsigned lds_addr, const void* gaddr) {
    asm volatile("global_load_async_to_lds_b128 %0, %1, off"
                 :: "v"(lds_addr), "v"((unsigned long long)(uintptr_t)gaddr)
                 : "memory");
}
__device__ __forceinline__ void wait_async0() {
    asm volatile("s_wait_asynccnt 0x0" ::: "memory");
}
__device__ __forceinline__ unsigned lds_off(const void* p) {
    return (unsigned)(uintptr_t)p;   // low 32 bits of flat addr == LDS byte offset
}

// ---------------------------------------------------------------------------
// Kernel 1: build per-channel 64x64 circulant matrix Mc from fft_filter.
// Exact equivalence: irfft2(rfft2(x)*F) == circular_conv(x, k) with
// k = irfft2(F_sym), F_sym = F with columns v=0 and v=4 u-symmetrized.
// ---------------------------------------------------------------------------
__global__ __launch_bounds__(64)
void build_mc(const float* __restrict__ F, unsigned short* __restrict__ Mc) {
    __shared__ float Fraw[40];
    __shared__ float k[64];
    const int c = blockIdx.x, t = threadIdx.x;
    if (t < 40) Fraw[t] = F[c * 40 + t];
    __syncthreads();
    const int p = t >> 3, q = t & 7;
    const float W8 = 0.78539816339744830962f; // 2*pi/8
    float s = 0.f;
    #pragma unroll
    for (int u = 0; u < 8; ++u) {
        const int um = (8 - u) & 7;
        const float f0 = 0.5f * (Fraw[u * 5 + 0] + Fraw[um * 5 + 0]);
        const float f4 = 0.5f * (Fraw[u * 5 + 4] + Fraw[um * 5 + 4]);
        s += f0 * __cosf(W8 * (float)(u * p));
        s += f4 * __cosf(W8 * (float)(u * p + 4 * q));
        #pragma unroll
        for (int v = 1; v <= 3; ++v)
            s += 2.f * Fraw[u * 5 + v] * __cosf(W8 * (float)(u * p + v * q));
    }
    k[t] = s * (1.0f / 64.0f);
    __syncthreads();
    // Row i = t (output pq). Mc[i][j] = k[(p-a)&7][(q-b)&7], j=(a,b).
    unsigned int* out = (unsigned int*)(Mc + (size_t)c * 4096 + t * 64);
    #pragma unroll
    for (int j2 = 0; j2 < 32; ++j2) {
        const int j0 = 2 * j2, j1 = j0 + 1;
        const float k0 = k[(((p - (j0 >> 3)) & 7) << 3) + ((q - (j0 & 7)) & 7)];
        const float k1 = k[(((p - (j1 >> 3)) & 7) << 3) + ((q - (j1 & 7)) & 7)];
        out[j2] = pack2(k0, k1);
    }
}

// ---------------------------------------------------------------------------
// Kernel 2: GEMM1  t[b][c][blk][pq] (bf16, block-major) = w_in(384x384) * x
// Tile: M=128 x N=128, K-step 32, 8 waves (4M x 2N), each wave 32Mx64N.
// ---------------------------------------------------------------------------
__global__ __launch_bounds__(256)
void gemm1(const float* __restrict__ x, const float* __restrict__ w_in,
           unsigned short* __restrict__ t) {
    __shared__ unsigned short lds[16384];          // A:[0,5120) B:[5120,10240); D reuse 32KB
    unsigned short* As = lds;
    unsigned short* Bs = lds + 5120;
    const int tid  = threadIdx.x;
    const int nt   = blockIdx.x, mt = blockIdx.y, b = blockIdx.z;
    const int m0   = mt * 128, N0 = nt * 128, blk0 = N0 >> 6;
    const int wave = tid >> 5, lane = tid & 31;
    const int wm   = (wave & 3) * 32, wn = (wave >> 2) * 64;
    const int lm   = lane & 15, kh = lane >> 4;

    v8f acc[2][4];
    #pragma unroll
    for (int i = 0; i < 2; ++i)
        #pragma unroll
        for (int j = 0; j < 4; ++j) acc[i][j] = (v8f){0,0,0,0,0,0,0,0};

    for (int k0 = 0; k0 < DIM; k0 += 32) {
        __syncthreads();
        // A: w_in[m0+m][k0+k] -> bf16 pairs (k-adjacent within float2)
        #pragma unroll
        for (int i = 0; i < 8; ++i) {
            const int e = tid + 256 * i;           // 2048 pairs
            const int k2 = e & 15, m = e >> 4;
            const float2 v = *(const float2*)(w_in + (size_t)(m0 + m) * DIM + k0 + 2 * k2);
            *(unsigned int*)(As + m * 40 + 2 * k2) = pack2(v.x, v.y);
        }
        // B: x, blocked pixel enumeration; two k-adjacent strided loads -> one b32 store
        #pragma unroll
        for (int i = 0; i < 8; ++i) {
            const int e = tid + 256 * i;           // 2048 (n, k-pair) pairs
            const int n = e & 127, k2 = e >> 7;    // k = 2*k2, 16 pairs
            const int blk = blk0 + (n >> 6);
            const int hb = blk >> 5, wb = blk & 31;
            const int p = (n >> 3) & 7, q = n & 7;
            const size_t pixoff = (size_t)(hb * 8 + p) * WSZ + wb * 8 + q;
            const float v0 = x[((size_t)(b * DIM + k0 + 2 * k2)     * HSZ) * WSZ + pixoff];
            const float v1 = x[((size_t)(b * DIM + k0 + 2 * k2 + 1) * HSZ) * WSZ + pixoff];
            *(unsigned int*)(Bs + n * 40 + 2 * k2) = pack2(v0, v1);
        }
        __syncthreads();
        Frag16 af[2], bf[4];
        #pragma unroll
        for (int i = 0; i < 2; ++i) {
            const unsigned short* r = As + (wm + i * 16 + lm) * 40;
            af[i].u[0] = *(const uint4*)(r + kh * 8);        // K kh*8..+7
            af[i].u[1] = *(const uint4*)(r + 16 + kh * 8);   // K 16+kh*8..+7
        }
        #pragma unroll
        for (int j = 0; j < 4; ++j) {
            const unsigned short* r = Bs + (wn + j * 16 + lm) * 40;
            bf[j].u[0] = *(const uint4*)(r + kh * 8);        // K kh*8..+7
            bf[j].u[1] = *(const uint4*)(r + 16 + kh * 8);   // K 16+kh*8..+7
        }
        #pragma unroll
        for (int i = 0; i < 2; ++i)
            #pragma unroll
            for (int j = 0; j < 4; ++j)
                acc[i][j] = wmma_bf16(af[i], bf[j], acc[i][j]);
    }
    // Epilogue: stage D tile (bf16) in LDS, then packed block-major stores.
    __syncthreads();
    #pragma unroll
    for (int i = 0; i < 2; ++i)
        #pragma unroll
        for (int j = 0; j < 4; ++j) {
            const int n = wn + j * 16 + lm;
            #pragma unroll
            for (int r2 = 0; r2 < 4; ++r2) {
                const unsigned int u = pack2(acc[i][j][2 * r2], acc[i][j][2 * r2 + 1]);
                lds[(wm + i * 16 + 2 * r2     + 8 * kh) * 128 + n] = (unsigned short)(u & 0xFFFFu);
                lds[(wm + i * 16 + 2 * r2 + 1 + 8 * kh) * 128 + n] = (unsigned short)(u >> 16);
            }
        }
    __syncthreads();
    {
        const int cc = tid >> 1, half = tid & 1;
        uint4* dst = (uint4*)(t + ((size_t)((b * HIDN + m0 + cc) * NBLK + blk0 + half)) * 64);
        const uint4* src = (const uint4*)(lds + cc * 128 + half * 64);
        #pragma unroll
        for (int i = 0; i < 8; ++i) dst[i] = src[i];
    }
}

// ---------------------------------------------------------------------------
// Kernel 3: per-(b,c) circulant filter: t_blk = Mc[c] (64x64) * t_blk, in place.
// Staging uses async global->LDS (ASYNCcnt) since layouts match contiguously.
// ---------------------------------------------------------------------------
__global__ __launch_bounds__(256)
void blockfilter(unsigned short* __restrict__ t, const unsigned short* __restrict__ Mc) {
    __shared__ unsigned short sA[64 * 64];     // Mc[c] row-major (8 KB)
    __shared__ unsigned short sB[128 * 64];    // 128 blocks (cols), 64 pq each (16 KB)
    const int bc = blockIdx.y;                 // b*384 + c
    const int c  = bc % HIDN;
    const int n0 = blockIdx.x * 128;
    const int tid = threadIdx.x, wave = tid >> 5, lane = tid & 31;
    {
        const char* srcA = (const char*)(Mc + (size_t)c * 4096);
        const char* srcB = (const char*)(t + ((size_t)bc * NBLK + n0) * 64);
        const unsigned ldsA = lds_off(sA) + tid * 16;
        const unsigned ldsB = lds_off(sB) + tid * 16;
        async_g2l_b128(ldsA,        srcA + tid * 16);
        async_g2l_b128(ldsA + 4096, srcA + tid * 16 + 4096);
        #pragma unroll
        for (int i = 0; i < 4; ++i)
            async_g2l_b128(ldsB + i * 4096, srcB + tid * 16 + i * 4096);
        wait_async0();
    }
    __syncthreads();
    const int wm = (wave & 3) * 16, wn = (wave >> 2) * 64;
    const int lm = lane & 15, kh = lane >> 4;
    v8f acc[4];
    #pragma unroll
    for (int j = 0; j < 4; ++j) acc[j] = (v8f){0,0,0,0,0,0,0,0};
    #pragma unroll
    for (int ks = 0; ks < 2; ++ks) {
        Frag16 af;
        const unsigned short* ra = sA + (wm + lm) * 64 + ks * 32;
        af.u[0] = *(const uint4*)(ra + kh * 8);
        af.u[1] = *(const uint4*)(ra + 16 + kh * 8);
        #pragma unroll
        for (int j = 0; j < 4; ++j) {
            Frag16 bf;
            const unsigned short* rb = sB + (wn + j * 16 + lm) * 64 + ks * 32;
            bf.u[0] = *(const uint4*)(rb + kh * 8);
            bf.u[1] = *(const uint4*)(rb + 16 + kh * 8);
            acc[j] = wmma_bf16(af, bf, acc[j]);
        }
    }
    #pragma unroll
    for (int j = 0; j < 4; ++j) {
        const int n = wn + j * 16 + lm;
        unsigned int tmp[4];
        #pragma unroll
        for (int r2 = 0; r2 < 4; ++r2)
            tmp[r2] = pack2(acc[j][2 * r2], acc[j][2 * r2 + 1]);
        *(uint4*)(t + ((size_t)bc * NBLK + n0 + n) * 64 + wm + 8 * kh) =
            *(const uint4*)tmp;
    }
}

// ---------------------------------------------------------------------------
// Kernel 4: depthwise 3x3 (cross-correlation, zero-pad SAME) + exact GELU gate.
// Reads block-major t2, writes row-major bf16 g[b][c<192][pix].
// ---------------------------------------------------------------------------
__global__ __launch_bounds__(256)
void dwgate(const unsigned short* __restrict__ t2, const float* __restrict__ wdw,
            unsigned short* __restrict__ g) {
    const int idx = blockIdx.x * 256 + threadIdx.x;   // 2*192*65536 total
    const int pix = idx & 65535;
    const int cc  = (idx >> 16) % 192;
    const int b   = idx / (192 * 65536);
    const int h = pix >> 8, w = pix & 255;
    const float* w1 = wdw + cc * 9;
    const float* w2 = wdw + (cc + 192) * 9;
    float a1 = 0.f, a2 = 0.f;
    #pragma unroll
    for (int dh = -1; dh <= 1; ++dh)
        #pragma unroll
        for (int dw = -1; dw <= 1; ++dw) {
            const int hh = h + dh, ww = w + dw;
            if (hh < 0 || hh > 255 || ww < 0 || ww > 255) continue;
            const int blk = ((hh >> 3) << 5) + (ww >> 3);
            const int pq  = ((hh & 7) << 3) + (ww & 7);
            const size_t o1 = ((size_t)(b * HIDN + cc)       * NBLK + blk) * 64 + pq;
            const size_t o2 = ((size_t)(b * HIDN + cc + 192) * NBLK + blk) * 64 + pq;
            const int wi = (dh + 1) * 3 + (dw + 1);
            a1 += bf2f(t2[o1]) * w1[wi];
            a2 += bf2f(t2[o2]) * w2[wi];
        }
    const float gl = 0.5f * a1 * (1.0f + erff(a1 * 0.70710678118654752f));
    g[((size_t)(b * 192 + cc)) * HW + pix] = f2bf(gl * a2);
}

// ---------------------------------------------------------------------------
// Kernel 5: GEMM2  out(fp32, row-major) = w_out(384x192) * g(192xHW), per batch.
// ---------------------------------------------------------------------------
__global__ __launch_bounds__(256)
void gemm2(const unsigned short* __restrict__ g, const float* __restrict__ w_out,
           float* __restrict__ out) {
    __shared__ unsigned short lds[10240];
    unsigned short* As = lds;
    unsigned short* Bs = lds + 5120;
    const int tid = threadIdx.x;
    const int nt = blockIdx.x, mt = blockIdx.y, b = blockIdx.z;
    const int m0 = mt * 128, N0 = nt * 128;
    const int wave = tid >> 5, lane = tid & 31;
    const int wm = (wave & 3) * 32, wn = (wave >> 2) * 64;
    const int lm = lane & 15, kh = lane >> 4;

    v8f acc[2][4];
    #pragma unroll
    for (int i = 0; i < 2; ++i)
        #pragma unroll
        for (int j = 0; j < 4; ++j) acc[i][j] = (v8f){0,0,0,0,0,0,0,0};

    for (int k0 = 0; k0 < 192; k0 += 32) {
        __syncthreads();
        #pragma unroll
        for (int i = 0; i < 8; ++i) {
            const int e = tid + 256 * i;
            const int k2 = e & 15, m = e >> 4;
            const float2 v = *(const float2*)(w_out + (size_t)(m0 + m) * 192 + k0 + 2 * k2);
            *(unsigned int*)(As + m * 40 + 2 * k2) = pack2(v.x, v.y);
        }
        #pragma unroll
        for (int i = 0; i < 8; ++i) {
            const int e = tid + 256 * i;
            const int n = e & 127, k2 = e >> 7;    // k = 2*k2
            const unsigned int u0 = g[((size_t)(b * 192 + k0 + 2 * k2))     * HW + N0 + n];
            const unsigned int u1 = g[((size_t)(b * 192 + k0 + 2 * k2 + 1)) * HW + N0 + n];
            *(unsigned int*)(Bs + n * 40 + 2 * k2) = u0 | (u1 << 16);
        }
        __syncthreads();
        Frag16 af[2], bf[4];
        #pragma unroll
        for (int i = 0; i < 2; ++i) {
            const unsigned short* r = As + (wm + i * 16 + lm) * 40;
            af[i].u[0] = *(const uint4*)(r + kh * 8);
            af[i].u[1] = *(const uint4*)(r + 16 + kh * 8);
        }
        #pragma unroll
        for (int j = 0; j < 4; ++j) {
            const unsigned short* r = Bs + (wn + j * 16 + lm) * 40;
            bf[j].u[0] = *(const uint4*)(r + kh * 8);
            bf[j].u[1] = *(const uint4*)(r + 16 + kh * 8);
        }
        #pragma unroll
        for (int i = 0; i < 2; ++i)
            #pragma unroll
            for (int j = 0; j < 4; ++j)
                acc[i][j] = wmma_bf16(af[i], bf[j], acc[i][j]);
    }
    #pragma unroll
    for (int i = 0; i < 2; ++i)
        #pragma unroll
        for (int j = 0; j < 4; ++j) {
            const int n = N0 + wn + j * 16 + lm;
            #pragma unroll
            for (int r = 0; r < 8; ++r) {
                const int m = m0 + wm + i * 16 + r + 8 * kh;
                out[((size_t)(b * DIM + m)) * HW + n] = acc[i][j][r];
            }
        }
}

extern "C" void kernel_launch(void* const* d_in, const int* in_sizes, int n_in,
                              void* d_out, int out_size, void* d_ws, size_t ws_size,
                              hipStream_t stream) {
    (void)in_sizes; (void)n_in; (void)out_size; (void)ws_size;
    const float* x    = (const float*)d_in[0];
    const float* fftF = (const float*)d_in[1];
    const float* w_in = (const float*)d_in[2];
    const float* w_dw = (const float*)d_in[3];
    const float* w_out= (const float*)d_in[4];
    float* out = (float*)d_out;

    unsigned short* t  = (unsigned short*)d_ws;                 // 2*384*HW bf16
    unsigned short* g  = t + (size_t)BATCH * HIDN * HW;         // 2*192*HW bf16
    unsigned short* Mc = g + (size_t)BATCH * 192 * HW;          // 384*4096 bf16

    build_mc<<<HIDN, 64, 0, stream>>>(fftF, Mc);
    gemm1<<<dim3(HW / 128, HIDN / 128, BATCH), 256, 0, stream>>>(x, w_in, t);
    blockfilter<<<dim3(NBLK / 128, BATCH * HIDN), 256, 0, stream>>>(t, Mc);
    dwgate<<<(BATCH * 192 * HW) / 256, 256, 0, stream>>>(t, w_dw, g);
    gemm2<<<dim3(HW / 128, DIM / 128, BATCH), 256, 0, stream>>>(g, w_out, out);
}
